// MultiHeadAttn_41686952575777
// MI455X (gfx1250) — matrix operand order repaired and checked by reading, not compiled, not run on hardware
//
#include <hip/hip_runtime.h>
#include <hip/hip_bf16.h>
#include <math.h>

typedef _Float16 half_t;
typedef __attribute__((ext_vector_type(16))) _Float16 v16h;
typedef __attribute__((ext_vector_type(8)))  _Float16 v8h;
typedef __attribute__((ext_vector_type(8)))  float    v8f;
typedef __attribute__((ext_vector_type(4)))  unsigned int v4u;
typedef __attribute__((ext_vector_type(8)))  int v8i;
typedef __attribute__((ext_vector_type(4)))  int v4i;

union V16U { v16h v; v8h h[2]; };

// ---- WMMA helper: D = A(16x32 f16) * B(32x16 f16) + C(16x16 f32) ----
__device__ __forceinline__ v8f wmma_f32_f16(v16h a, v16h b, v8f c) {
    return __builtin_amdgcn_wmma_f32_16x16x32_f16(
        /*neg_a=*/false, a, /*neg_b=*/false, b,
        /*c_mod=*/(short)0, c, /*reuse_a=*/false, /*reuse_b=*/false);
}

// A-fragment (16x32, 16-bit) from LDS tile with row stride `ldk` halves.
// Lane L -> M = L%16; halves 0..7 hold K = g*8..g*8+7,
// halves 8..15 hold K = 16+g*8..16+g*8+7, where g = L/16.
__device__ __forceinline__ v16h load_a_frag(const half_t* base, int ldk, int lane) {
    int m = lane & 15;
    int g = lane >> 4;
    const half_t* row = base + m * ldk;
    V16U u;
    u.h[0] = *reinterpret_cast<const v8h*>(row + g * 8);
    u.h[1] = *reinterpret_cast<const v8h*>(row + 16 + g * 8);
    return u.v;
}

// B-fragment (32x16, 16-bit) from an LDS tile stored TRANSPOSED as [n][k].
__device__ __forceinline__ v16h load_b_frag(const half_t* baseT, int ldk, int lane) {
    int n = lane & 15;
    int g = lane >> 4;
    const half_t* row = baseT + n * ldk;
    V16U u;
    u.h[0] = *reinterpret_cast<const v8h*>(row + g * 8);
    u.h[1] = *reinterpret_cast<const v8h*>(row + 16 + g * 8);
    return u.v;
}

// ---- Tensor Data Mover: 2D f16 tile (rows x cols) global -> LDS ----
// D# per CDNA5 ISA §8: group0 = {flags, lds_addr, global_addr, type=2},
// group1 = {data_size=2B, tensor_dim0/1, tile_dim0/1, dim0_stride}.
// amdgpu-toolchain (clang-23) 6-arg builtin form.
__device__ __forceinline__ void tdm_load_2d_f16(unsigned lds_byte_off,
                                                const half_t* gptr,
                                                unsigned rows, unsigned cols,
                                                unsigned row_stride_elems) {
    unsigned long long ga = (unsigned long long)(size_t)gptr;
    v4u g0;
    g0[0] = 1u;                                          // count=1, user D#
    g0[1] = lds_byte_off;                                // lds_addr
    g0[2] = (unsigned)(ga & 0xFFFFFFFFu);                // global_addr[31:0]
    g0[3] = (unsigned)((ga >> 32) & 0x01FFFFFFu) | (2u << 30); // [56:32] | type=2
    v8i g1;
    g1[0] = (int)(1u << 16);                             // data_size code 1 = 2 bytes
    g1[1] = (int)((cols & 0xFFFFu) << 16);               // tensor_dim0[15:0]
    g1[2] = (int)((rows & 0xFFFFu) << 16);               // tensor_dim1[15:0]
    g1[3] = (int)((cols & 0xFFFFu) << 16);               // tile_dim0
    g1[4] = (int)(rows & 0xFFFFu);                       // tile_dim1 (tile_dim2=0)
    g1[5] = (int)row_stride_elems;                       // tensor_dim0_stride[31:0]
    g1[6] = 0;
    g1[7] = 0;
    v4i z4 = {0, 0, 0, 0};                               // 2D: groups 2/3 unused
    v8i z8 = {0, 0, 0, 0, 0, 0, 0, 0};
    __builtin_amdgcn_tensor_load_to_lds(g0, g1, z4, z4, z8, 0);
}

__device__ __forceinline__ unsigned lds_offset_of(const void* p) {
    return (unsigned)(size_t)p;   // LDS offset is the low 32 bits of the generic address
}

// =====================================================================
// Kernel 1: per-head input projection  Y[m, h*64+c] = sum_d X[m,d]*W[h,d,c]
// X: [M,1024] f32, W: [16,1024,64] f32, Y: [M,1024] f16
// Block tile 128x64, 128 threads (4 waves, 32 rows each), K step 32.
// =====================================================================
__global__ __launch_bounds__(128)
void mha_proj_qkv(const float* __restrict__ X, const float* __restrict__ W,
                  half_t* __restrict__ Y) {
    __shared__ half_t As[128][32];  // [m][k]
    __shared__ half_t Wt[64][32];   // [n][k]  (B transposed for B-frag reads)

    const int tid  = threadIdx.x;
    const int lane = tid & 31;
    const int wave = tid >> 5;
    const int m0   = blockIdx.x * 128;
    const int n0   = blockIdx.y * 64;

    v8f acc[2][4] = {};

    for (int k0 = 0; k0 < 1024; k0 += 32) {
        // ---- stage A tile: 128 rows x 32 f32 -> f16 (1024 float4 / 128 thr) ----
        #pragma unroll
        for (int i = 0; i < 8; ++i) {
            int idx = tid + i * 128;          // 0..1023
            int r   = idx >> 3;
            int c4  = idx & 7;
            float4 f = reinterpret_cast<const float4*>(X + (size_t)(m0 + r) * 1024 + k0)[c4];
            half_t* dst = &As[r][c4 * 4];
            dst[0] = (half_t)f.x; dst[1] = (half_t)f.y;
            dst[2] = (half_t)f.z; dst[3] = (half_t)f.w;
        }
        // ---- stage W tile transposed: Wt[n][k] = W[h, k0+k, col] ----
        #pragma unroll
        for (int i = 0; i < 16; ++i) {
            int idx = tid * 16 + i;           // 0..2047, k fastest
            int n   = idx >> 5;
            int k   = idx & 31;
            int ng  = n0 + n;
            int h   = ng >> 6;
            int col = ng & 63;
            Wt[n][k] = (half_t)W[((size_t)h * 1024 + (k0 + k)) * 64 + col];
        }
        __syncthreads();

        v16h af0 = load_a_frag(&As[wave * 32][0],      32, lane);
        v16h af1 = load_a_frag(&As[wave * 32 + 16][0], 32, lane);
        #pragma unroll
        for (int nt = 0; nt < 4; ++nt) {
            v16h bf = load_b_frag(&Wt[nt * 16][0], 32, lane);
            acc[0][nt] = wmma_f32_f16(af0, bf, acc[0][nt]);
            acc[1][nt] = wmma_f32_f16(af1, bf, acc[1][nt]);
        }
        __syncthreads();
    }

    // ---- store f16, C-layout: VGPR r -> row r + 8*(lane/16), col lane%16 ----
    const int g  = lane >> 4;
    const int nl = lane & 15;
    #pragma unroll
    for (int mi = 0; mi < 2; ++mi)
        #pragma unroll
        for (int nt = 0; nt < 4; ++nt)
            #pragma unroll
            for (int r = 0; r < 8; ++r) {
                int m = m0 + wave * 32 + mi * 16 + r + 8 * g;
                int n = n0 + nt * 16 + nl;
                Y[(size_t)m * 1024 + n] = (half_t)acc[mi][nt][r];
            }
}

// =====================================================================
// Kernel 2: flash attention per (b, h, 64-row q-block).
// Q and K tiles arrive via the Tensor Data Mover (TENSORcnt); V is staged
// manually because it needs a transpose. grid = (L/64, B*H), block = 128.
// =====================================================================
__global__ __launch_bounds__(128)
void mha_flash_attn(const half_t* __restrict__ Qh, const half_t* __restrict__ Kh,
                    const half_t* __restrict__ Vh, half_t* __restrict__ O) {
    __shared__ half_t Qs[64][64];      // [q][dk]
    __shared__ half_t Ks[64][64];      // [s][dk]   (== B-transposed for QK^T)
    __shared__ half_t Vt[64][64];      // [dv][s]   (B-transposed for PV)
    __shared__ half_t Ps[4][16][64];   // per-wave P staging [q][s]

    const int tid  = threadIdx.x;
    const int lane = tid & 31;
    const int wave = tid >> 5;
    const int g    = lane >> 4;
    const int nl   = lane & 15;

    const int b  = blockIdx.y & 3;
    const int h  = blockIdx.y >> 2;
    const int q0 = blockIdx.x * 64;

    const half_t* Qbase = Qh + ((size_t)(b * 2048 + q0)) * 1024 + h * 64;
    const half_t* Kbase = Kh + ((size_t)(b * 2048)) * 1024 + h * 64;
    const half_t* Vbase = Vh + ((size_t)(b * 2048)) * 1024 + h * 64;

    // ---- Q tile once via TDM (wave 0 issues; barrier publishes) ----
    if (wave == 0) {
        tdm_load_2d_f16(lds_offset_of(&Qs[0][0]), Qbase, 64, 64, 1024);
        __builtin_amdgcn_s_wait_tensorcnt(0);
    }

    float m_run[8], l_run[8];
    #pragma unroll
    for (int r = 0; r < 8; ++r) { m_run[r] = -INFINITY; l_run[r] = 0.0f; }
    v8f o[4] = {};

    const float scale = 0.03125f;   // 1/sqrt(1024)

    for (int s0 = 0; s0 < 2048; s0 += 64) {
        // ---- K tile via TDM; V tile staged + transposed by all waves ----
        if (wave == 0)
            tdm_load_2d_f16(lds_offset_of(&Ks[0][0]),
                            Kbase + (size_t)s0 * 1024, 64, 64, 1024);
        #pragma unroll
        for (int i = 0; i < 4; ++i) {
            int idx = tid + i * 128;
            int r = idx >> 3, c = (idx & 7) * 8;
            v8h vv = *reinterpret_cast<const v8h*>(Vbase + (size_t)(s0 + r) * 1024 + c);
            #pragma unroll
            for (int j = 0; j < 8; ++j) Vt[c + j][r] = vv[j];
        }
        if (s0 + 64 < 2048)  // hint next V block toward the caches
            __builtin_prefetch(Vbase + (size_t)(s0 + 64 + (tid >> 1)) * 1024 + (tid & 1) * 32, 0, 1);
        if (wave == 0)
            __builtin_amdgcn_s_wait_tensorcnt(0);
        __syncthreads();

        // ---- scores: S(16x64) = Q(16x64) * K^T ----
        v8f sc[4] = {};
        #pragma unroll
        for (int kk = 0; kk < 64; kk += 32) {
            v16h af = load_a_frag(&Qs[wave * 16][kk], 64, lane);
            #pragma unroll
            for (int nt = 0; nt < 4; ++nt) {
                v16h bf = load_b_frag(&Ks[nt * 16][kk], 64, lane);
                sc[nt] = wmma_f32_f16(af, bf, sc[nt]);
            }
        }

        // ---- online softmax; lane holds rows r+8g, col nl (+16*nt) ----
        #pragma unroll
        for (int r = 0; r < 8; ++r) {
            float mx = -INFINITY;
            #pragma unroll
            for (int nt = 0; nt < 4; ++nt) mx = fmaxf(mx, sc[nt][r] * scale);
            #pragma unroll
            for (int off = 1; off < 16; off <<= 1)
                mx = fmaxf(mx, __shfl_xor(mx, off, 32));

            float m_new = fmaxf(m_run[r], mx);
            float corr  = __expf(m_run[r] - m_new);
            float psum  = 0.0f;
            #pragma unroll
            for (int nt = 0; nt < 4; ++nt) {
                float p = __expf(sc[nt][r] * scale - m_new);
                psum += p;
                Ps[wave][r + 8 * g][nt * 16 + nl] = (half_t)p;
            }
            #pragma unroll
            for (int off = 1; off < 16; off <<= 1)
                psum += __shfl_xor(psum, off, 32);

            l_run[r] = l_run[r] * corr + psum;
            m_run[r] = m_new;
            #pragma unroll
            for (int nt = 0; nt < 4; ++nt) o[nt][r] *= corr;
        }

        // same-wave LDS RAW on Ps before A-fragment readback
        asm volatile("s_wait_dscnt 0" ::: "memory");

        // ---- O += P(16x64) * V(64x64) ----
        #pragma unroll
        for (int kk = 0; kk < 64; kk += 32) {
            v16h af = load_a_frag(&Ps[wave][0][kk], 64, lane);
            #pragma unroll
            for (int nt = 0; nt < 4; ++nt) {
                v16h bf = load_b_frag(&Vt[nt * 16][kk], 64, lane);
                o[nt] = wmma_f32_f16(af, bf, o[nt]);
            }
        }
        __syncthreads();   // protect Ks/Vt before next iteration's staging
    }

    // ---- epilogue: normalize and store f16 head-output ----
    #pragma unroll
    for (int nt = 0; nt < 4; ++nt)
        #pragma unroll
        for (int r = 0; r < 8; ++r) {
            int row = q0 + wave * 16 + r + 8 * g;
            float inv = 1.0f / l_run[r];
            O[((size_t)(b * 2048 + row)) * 1024 + h * 64 + nt * 16 + nl] =
                (half_t)(o[nt][r] * inv);
        }
}

// =====================================================================
// Kernel 3: output projection  out[m,n] = sum_j A[m,j]*proj_w[n,j] + b[n]
// A: [8192,1024] f16, proj_w: [1024,1024] f32 (row n contiguous in j)
// Block tile 128x64, 4 waves x 32 rows.
// =====================================================================
__global__ __launch_bounds__(128)
void mha_out_proj(const half_t* __restrict__ A, const float* __restrict__ W,
                  const float* __restrict__ bias, float* __restrict__ out) {
    __shared__ half_t As[128][32];
    __shared__ half_t Wt[64][32];   // [n][k], contiguous-k reads of proj_w row n

    const int tid  = threadIdx.x;
    const int lane = tid & 31;
    const int wave = tid >> 5;
    const int m0   = blockIdx.x * 128;
    const int n0   = blockIdx.y * 64;

    v8f acc[2][4] = {};

    for (int k0 = 0; k0 < 1024; k0 += 32) {
        // A tile: 128x32 halves = 512 v8h chunks
        #pragma unroll
        for (int i = 0; i < 4; ++i) {
            int idx = tid + i * 128;
            int r = idx >> 2, c = (idx & 3) * 8;
            *reinterpret_cast<v8h*>(&As[r][c]) =
                *reinterpret_cast<const v8h*>(A + (size_t)(m0 + r) * 1024 + k0 + c);
        }
        // W tile: Wt[n][k] = proj_w[(n0+n)*1024 + k0+k]  (coalesced float4)
        #pragma unroll
        for (int i = 0; i < 4; ++i) {
            int idx = tid + i * 128;          // 0..511
            int n   = idx >> 3;
            int c4  = idx & 7;
            float4 f = reinterpret_cast<const float4*>(W + (size_t)(n0 + n) * 1024 + k0)[c4];
            half_t* dst = &Wt[n][c4 * 4];
            dst[0] = (half_t)f.x; dst[1] = (half_t)f.y;
            dst[2] = (half_t)f.z; dst[3] = (half_t)f.w;
        }
        __syncthreads();

        v16h af0 = load_a_frag(&As[wave * 32][0],      32, lane);
        v16h af1 = load_a_frag(&As[wave * 32 + 16][0], 32, lane);
        #pragma unroll
        for (int nt = 0; nt < 4; ++nt) {
            v16h bf = load_b_frag(&Wt[nt * 16][0], 32, lane);
            acc[0][nt] = wmma_f32_f16(af0, bf, acc[0][nt]);
            acc[1][nt] = wmma_f32_f16(af1, bf, acc[1][nt]);
        }
        __syncthreads();
    }

    const int g  = lane >> 4;
    const int nl = lane & 15;
    #pragma unroll
    for (int mi = 0; mi < 2; ++mi)
        #pragma unroll
        for (int nt = 0; nt < 4; ++nt)
            #pragma unroll
            for (int r = 0; r < 8; ++r) {
                int m = m0 + wave * 32 + mi * 16 + r + 8 * g;
                int n = n0 + nt * 16 + nl;
                out[(size_t)m * 1024 + n] = acc[mi][nt][r] + bias[n];
            }
}

// =====================================================================
extern "C" void kernel_launch(void* const* d_in, const int* in_sizes, int n_in,
                              void* d_out, int out_size, void* d_ws, size_t ws_size,
                              hipStream_t stream) {
    (void)in_sizes; (void)n_in; (void)out_size; (void)ws_size;

    const float* q      = (const float*)d_in[0];
    const float* k      = (const float*)d_in[1];
    const float* v      = (const float*)d_in[2];
    const float* w_q    = (const float*)d_in[3];
    const float* w_k    = (const float*)d_in[4];
    const float* w_v    = (const float*)d_in[5];
    const float* proj_w = (const float*)d_in[6];
    const float* proj_b = (const float*)d_in[7];
    float*       out    = (float*)d_out;

    const size_t MN = (size_t)8192 * 1024;   // B*L x H*D
    half_t* qh = (half_t*)d_ws;
    half_t* kh = qh + MN;
    half_t* vh = kh + MN;
    half_t* ao = vh + MN;

    dim3 blk(128);
    dim3 gGemm(64, 16);                      // 8192/128 x 1024/64

    mha_proj_qkv<<<gGemm, blk, 0, stream>>>(q, w_q, qh);
    mha_proj_qkv<<<gGemm, blk, 0, stream>>>(k, w_k, kh);
    mha_proj_qkv<<<gGemm, blk, 0, stream>>>(v, w_v, vh);

    mha_flash_attn<<<dim3(32, 64), blk, 0, stream>>>(qh, kh, vh, ao);

    mha_out_proj<<<gGemm, blk, 0, stream>>>(ao, proj_w, proj_b, out);
}
